// SiameseModel_60859686585078
// MI455X (gfx1250) — matrix operand order repaired and checked by reading, not compiled, not run on hardware
//
#include <hip/hip_runtime.h>
#include <stdint.h>

typedef __bf16 v16bf __attribute__((ext_vector_type(16)));
typedef float  v8f   __attribute__((ext_vector_type(8)));
typedef unsigned short u16b;

#define B_   128
#define T_   512
#define F_   128
#define H_   512
#define NH_  256
#define H2_  1024
#define K1_  640    /* F + H   (layer1 fused K) */
#define K2_  1024   /* 2H      (layer2 fused K) */

/* ---------------- workspace layout (bytes) ---------------- */
static constexpr size_t OFF_WU1T = 0;                               /* [2][512][640]  bf16 */
static constexpr size_t SZ_WU1T  = (size_t)2*512*640*2;
static constexpr size_t OFF_WU2T = OFF_WU1T + SZ_WU1T;              /* [2][512][1024] bf16 */
static constexpr size_t SZ_WU2T  = (size_t)2*512*1024*2;
static constexpr size_t OFF_WDT  = OFF_WU2T + SZ_WU2T;              /* [256][1024]    bf16 */
static constexpr size_t SZ_WDT   = (size_t)256*1024*2;
static constexpr size_t OFF_XB   = OFF_WDT + SZ_WDT;                /* [2][T][B][F]   bf16 */
static constexpr size_t SZ_XB    = (size_t)2*T_*B_*F_*2;
static constexpr size_t OFF_MASK = OFF_XB + SZ_XB;                  /* [2][T][B]      f32  */
static constexpr size_t SZ_MASK  = (size_t)2*T_*B_*4;
static constexpr size_t OFF_HFIN = OFF_MASK + SZ_MASK;              /* [4][B][H]      f32  */
static constexpr size_t SZ_HFIN  = (size_t)4*B_*H_*4;
static constexpr size_t OFF_HN   = OFF_HFIN + SZ_HFIN;              /* [2][B][2H]     bf16 */
static constexpr size_t SZ_HN    = (size_t)2*B_*H2_*2;
static constexpr size_t OFF_FS   = OFF_HN + SZ_HN;                  /* [2][B][NH]     f32  */

/* ---------------- helpers ---------------- */
__device__ __forceinline__ u16b f2bf(float f) {
  uint32_t u = __float_as_uint(f);
  uint32_t r = u + 0x7FFFu + ((u >> 16) & 1u);   /* RNE */
  return (u16b)(r >> 16);
}
__device__ __forceinline__ float bf2f(u16b h) {
  return __uint_as_float(((uint32_t)h) << 16);
}

/* A-operand tile (16x32 MxK, bf16): lane half 0 holds K {k0..k0+7, k0+16..k0+23},
 * half 1 holds {k0+8.., k0+24..}; p must point at row's (k0 + half*8). */
__device__ __forceinline__ v16bf load_a(const u16b* p) {
  union { v16bf v; int4 q[2]; } u;
  u.q[0] = *(const int4*)(p);
  u.q[1] = *(const int4*)(p + 16);
  return u.v;
}
/* B-operand tile (32x16 KxN, bf16): lane half holds K {k0+half*8..+7, k0+16+half*8..+7}; p at k0+half*8. */
__device__ __forceinline__ v16bf load_b(const u16b* p) {
  union { v16bf v; int4 q[2]; } u;
  u.q[0] = *(const int4*)(p);
  u.q[1] = *(const int4*)(p + 16);
  return u.v;
}
__device__ __forceinline__ v8f wmma_bf16(v16bf a, v16bf b, v8f c) {
  return __builtin_amdgcn_wmma_f32_16x16x32_bf16(false, a, false, b, (short)0, c,
                                                 false, false);
}

/* one K-chunk: 8 M-tiles against one B tile, batched 4 loads + 4 WMMAs to
 * amortize the WMMA->VALU WAR hazard window (ISA 7.12.1). aStride in elems. */
__device__ __forceinline__ void gemm_chunk8(const u16b* abase, int aStride,
                                            v16bf bt, v8f* acc) {
#pragma unroll
  for (int g = 0; g < 2; ++g) {
    v16bf at[4];
#pragma unroll
    for (int q = 0; q < 4; ++q)
      at[q] = load_a(abase + (size_t)(g * 4 + q) * 16 * aStride);
#pragma unroll
    for (int q = 0; q < 4; ++q)
      acc[g * 4 + q] = wmma_bf16(at[q], bt, acc[g * 4 + q]);
  }
}

/* ---------------- prep: weight transpose+convert to bf16 [n][k] ---------------- */
__global__ void k_transpose_bf16(const float* __restrict__ src, int K, int N,
                                 u16b* __restrict__ dst, int dstStride, int k0) {
  int tid = blockIdx.x * blockDim.x + threadIdx.x;
  int total = K * N;
  if (tid >= total) return;
  int k = tid / N, n = tid % N;
  dst[(size_t)n * dstStride + k0 + k] = f2bf(src[(size_t)k * N + n]);
}

/* ---------------- prep: x -> bf16 (T,B,F) + mask (T,B) ---------------- */
__global__ void k_xprep(const float* __restrict__ x1, const float* __restrict__ x2,
                        u16b* __restrict__ xb, float* __restrict__ maskf) {
  __shared__ int anyflag;
  int i = blockIdx.y;
  int t = blockIdx.x / B_;
  int b = blockIdx.x % B_;
  const float* xi = i ? x2 : x1;
  float v = xi[((size_t)b * T_ + t) * F_ + threadIdx.x];
  xb[((size_t)i * T_ * B_ + (size_t)t * B_ + b) * F_ + threadIdx.x] = f2bf(v);
  if (threadIdx.x == 0) anyflag = 0;
  __syncthreads();
  if (v != 0.0f) atomicOr(&anyflag, 1);
  __syncthreads();
  if (threadIdx.x == 0)
    maskf[(size_t)i * T_ * B_ + (size_t)t * B_ + b] = anyflag ? 1.0f : 0.0f;
}

/* ---------------- async x-tile prefetch: global -> LDS (32KB, 2048 x b128) ------- */
__device__ __forceinline__ void issue_async_x(int tid, uint32_t ldsBase, const u16b* g) {
  uint32_t c0 = (uint32_t)tid * 2u;
#pragma unroll
  for (int k = 0; k < 2; ++k) {
    uint32_t lo = ldsBase + (c0 + k) * 16u;
    const u16b* gp = g + (size_t)(c0 + k) * 8;
    asm volatile("global_load_async_to_lds_b128 %0, %1, off"
                 :: "v"(lo), "v"((unsigned long long)(uintptr_t)gp) : "memory");
  }
}

/* ---------------- the recurrence: 4 persistent blocks (input x direction) -------- */
__global__ void __launch_bounds__(1024, 1)
k_rnn(char* __restrict__ ws,
      const float* __restrict__ bf1, const float* __restrict__ bf2,
      const float* __restrict__ bb1, const float* __restrict__ bb2) {
  extern __shared__ char smem[];
  u16b* HH  = (u16b*)smem;                       /* [128][1024] bf16 = 256KB: h1|h2 */
  const uint32_t XBUF_OFF = 128u * 1024u * 2u;   /* 262144 */
  u16b* XB0 = (u16b*)(smem + XBUF_OFF);          /* [2][128][128] bf16 = 64KB       */

  const int c = blockIdx.x;          /* combo */
  const int i = c >> 1, d = c & 1;   /* input, direction (0=fwd,1=bwd) */
  const u16b*  WU1t  = (const u16b*)(ws + OFF_WU1T) + (size_t)d * 512 * K1_;
  const u16b*  WU2t  = (const u16b*)(ws + OFF_WU2T) + (size_t)d * 512 * K2_;
  const u16b*  xbi   = (const u16b*)(ws + OFF_XB)   + (size_t)i * T_ * B_ * F_;
  const float* maskb = (const float*)(ws + OFF_MASK) + (size_t)i * T_ * B_;
  const float* b1 = d ? bb1 : bf1;
  const float* b2 = d ? bb2 : bf2;
  float* hfin = (float*)(ws + OFF_HFIN) + (size_t)c * B_ * H_;

  const int tid  = threadIdx.x;
  const int wave = tid >> 5;
  const int lane = tid & 31;
  const int half = lane >> 4;
  const int lr   = lane & 15;
  const int n0   = wave * 16;        /* 32 waves cover N = 512 */

  /* zero hidden state */
  {
    int4 z = make_int4(0, 0, 0, 0);
    int4* hq = (int4*)HH;
#pragma unroll 1
    for (int k = tid; k < (128 * 1024 * 2) / 16; k += 1024) hq[k] = z;
  }
  /* prefetch x for step 0 into buffer 0 */
  {
    int tt0 = d ? (T_ - 1) : 0;
    issue_async_x(tid, XBUF_OFF, xbi + (size_t)tt0 * B_ * F_);
  }
  asm volatile("s_wait_asynccnt 0" ::: "memory");
  __syncthreads();

  const u16b* wu1row = WU1t + (size_t)(n0 + lr) * K1_ + half * 8;
  const u16b* wu2row = WU2t + (size_t)(n0 + lr) * K2_ + half * 8;

#pragma unroll 1
  for (int t = 0; t < T_; ++t) {
    const int tt  = d ? (T_ - 1 - t) : t;
    const int buf = t & 1;
    if (t + 1 < T_) {
      int ttn = d ? (T_ - 2 - t) : (t + 1);
      issue_async_x(tid, XBUF_OFF + (uint32_t)(buf ^ 1) * 32768u,
                    xbi + (size_t)ttn * B_ * F_);
    }
    const float* mk = maskb + (size_t)tt * B_;

    /* ===== layer 1: out(128x512) = [x_t | h1](128x640) @ WU1t^T ===== */
    v8f acc[8];
#pragma unroll
    for (int mt = 0; mt < 8; ++mt) acc[mt] = v8f{};

    /* x part of K (kc 0..3): A from async-staged LDS x buffer */
    const u16b* xrow = XB0 + buf * (B_ * F_) + lr * F_ + half * 8;
#pragma unroll 1
    for (int kc = 0; kc < F_ / 32; ++kc) {
      const int k0 = kc * 32;
      v16bf bt = load_b(wu1row + k0);
      gemm_chunk8(xrow + k0, F_, bt, acc);
    }
    /* h1 part of K (kc 0..15): A from LDS-resident hidden state */
    const u16b* hrow = HH + lr * H2_ + half * 8;
#pragma unroll 1
    for (int kc = 0; kc < H_ / 32; ++kc) {
      const int k0 = kc * 32;
      v16bf bt = load_b(wu1row + F_ + k0);
      gemm_chunk8(hrow + k0, H2_, bt, acc);
    }
    __syncthreads();   /* all waves done reading old h1 */
    {
      float bn = b1[n0 + lr];
#pragma unroll
      for (int mt = 0; mt < 8; ++mt)
#pragma unroll
        for (int r = 0; r < 8; ++r) {
          int   M   = mt * 16 + r + half * 8;
          float hv  = tanhf(acc[mt][r] + bn);
          float m   = mk[M];
          int   idx = M * H2_ + (n0 + lr);
          float ov  = bf2f(HH[idx]);
          HH[idx] = f2bf(m > 0.5f ? hv : ov);
        }
    }
    __syncthreads();   /* new h1 visible */

    /* ===== layer 2: out(128x512) = [h1 | h2](128x1024) @ WU2t^T ===== */
#pragma unroll
    for (int mt = 0; mt < 8; ++mt) acc[mt] = v8f{};
#pragma unroll 1
    for (int kc = 0; kc < K2_ / 32; ++kc) {
      const int k0 = kc * 32;
      v16bf bt = load_b(wu2row + k0);
      gemm_chunk8(hrow + k0, H2_, bt, acc);
    }
    __syncthreads();   /* all waves done reading old h2 */
    {
      float bn = b2[n0 + lr];
#pragma unroll
      for (int mt = 0; mt < 8; ++mt)
#pragma unroll
        for (int r = 0; r < 8; ++r) {
          int   M   = mt * 16 + r + half * 8;
          float hv  = tanhf(acc[mt][r] + bn);
          float m   = mk[M];
          int   idx = M * H2_ + 512 + (n0 + lr);
          float ov  = bf2f(HH[idx]);
          HH[idx] = f2bf(m > 0.5f ? hv : ov);
        }
    }
    asm volatile("s_wait_asynccnt 0" ::: "memory");  /* my prefetch for t+1 done */
    __syncthreads();                                 /* everyone's prefetch done */
  }

  /* dump final h2 (B x H) as f32 */
#pragma unroll 1
  for (int e = tid; e < B_ * H_; e += 1024) {
    int b = e >> 9, h = e & (H_ - 1);
    hfin[e] = bf2f(HH[b * H2_ + 512 + h]);
  }
}

/* ---------------- batchnorm over batch + normalize to bf16 ---------------- */
__global__ void k_bn(char* __restrict__ ws,
                     const float* __restrict__ gamma, const float* __restrict__ beta) {
  int j = blockIdx.x * blockDim.x + threadIdx.x;
  if (j >= 2 * H2_) return;
  int i = j >> 10, col = j & (H2_ - 1);
  int dd = col >> 9, hcol = col & (H_ - 1);
  const float* hf = (const float*)(ws + OFF_HFIN) + (size_t)(i * 2 + dd) * B_ * H_ + hcol;
  float s = 0.f, ss = 0.f;
  for (int b = 0; b < B_; ++b) { float v = hf[(size_t)b * H_]; s += v; ss += v * v; }
  float mu  = s * (1.0f / B_);
  float var = ss * (1.0f / B_) - mu * mu;
  float sc  = rsqrtf(var + 1e-3f) * gamma[col];
  float sh  = beta[col] - mu * sc;
  u16b* hn = (u16b*)(ws + OFF_HN) + (size_t)i * B_ * H2_ + col;
  for (int b = 0; b < B_; ++b) hn[(size_t)b * H2_] = f2bf(hf[(size_t)b * H_] * sc + sh);
}

/* ---------------- dense (128x256x1024) WMMA + sigmoid ---------------- */
__global__ void __launch_bounds__(512, 1)
k_dense(char* __restrict__ ws, const float* __restrict__ bd) {
  const int i    = blockIdx.x;
  const int tid  = threadIdx.x;
  const int wave = tid >> 5;
  const int lane = tid & 31;
  const int half = lane >> 4;
  const int lr   = lane & 15;
  const int n0   = wave * 16;   /* 16 waves cover NH = 256 */

  const u16b* hn  = (const u16b*)(ws + OFF_HN) + (size_t)i * B_ * H2_;
  const u16b* wdt = (const u16b*)(ws + OFF_WDT);
  float*      fs  = (float*)(ws + OFF_FS) + (size_t)i * B_ * NH_;

  const u16b* wrow = wdt + (size_t)(n0 + lr) * K2_ + half * 8;
  const u16b* arow = hn + (size_t)lr * H2_ + half * 8;

  v8f acc[8];
#pragma unroll
  for (int mt = 0; mt < 8; ++mt) acc[mt] = v8f{};
#pragma unroll 1
  for (int kc = 0; kc < K2_ / 32; ++kc) {
    const int k0 = kc * 32;
    v16bf bt = load_b(wrow + k0);
    gemm_chunk8(arow + k0, H2_, bt, acc);
  }
  float bn = bd[n0 + lr];
#pragma unroll
  for (int mt = 0; mt < 8; ++mt)
#pragma unroll
    for (int r = 0; r < 8; ++r) {
      int   M = mt * 16 + r + half * 8;
      float v = acc[mt][r] + bn;
      fs[(size_t)M * NH_ + n0 + lr] = 1.0f / (1.0f + expf(-v));
    }
}

/* ---------------- cosine similarity of L2-normalized rows ---------------- */
__global__ void k_cos(char* __restrict__ ws, float* __restrict__ out) {
  int b = blockIdx.x;
  int lane = threadIdx.x;
  const float* f1 = (const float*)(ws + OFF_FS) + (size_t)b * NH_;
  const float* f2 = (const float*)(ws + OFF_FS) + (size_t)B_ * NH_ + (size_t)b * NH_;
  float dp = 0.f, a = 0.f, c = 0.f;
  for (int n = lane; n < NH_; n += 32) {
    float x = f1[n], y = f2[n];
    dp += x * y; a += x * x; c += y * y;
  }
#pragma unroll
  for (int o = 16; o > 0; o >>= 1) {
    dp += __shfl_xor(dp, o, 32);
    a  += __shfl_xor(a,  o, 32);
    c  += __shfl_xor(c,  o, 32);
  }
  if (lane == 0)
    out[b] = dp * rsqrtf(fmaxf(a, 1e-12f)) * rsqrtf(fmaxf(c, 1e-12f));
}

/* ---------------- host launch ---------------- */
extern "C" void kernel_launch(void* const* d_in, const int* in_sizes, int n_in,
                              void* d_out, int out_size, void* d_ws, size_t ws_size,
                              hipStream_t stream) {
  (void)in_sizes; (void)n_in; (void)out_size; (void)ws_size;
  char* ws = (char*)d_ws;
  const float* x1  = (const float*)d_in[0];
  const float* x2  = (const float*)d_in[1];
  const float* Wf1 = (const float*)d_in[2];
  const float* Uf1 = (const float*)d_in[3];
  const float* bf1 = (const float*)d_in[4];
  const float* Wf2 = (const float*)d_in[5];
  const float* Uf2 = (const float*)d_in[6];
  const float* bf2 = (const float*)d_in[7];
  const float* Wb1 = (const float*)d_in[8];
  const float* Ub1 = (const float*)d_in[9];
  const float* bb1 = (const float*)d_in[10];
  const float* Wb2 = (const float*)d_in[11];
  const float* Ub2 = (const float*)d_in[12];
  const float* bb2 = (const float*)d_in[13];
  const float* gamma = (const float*)d_in[14];
  const float* beta  = (const float*)d_in[15];
  const float* Wd  = (const float*)d_in[16];
  const float* bd  = (const float*)d_in[17];

  u16b* WU1t_f = (u16b*)(ws + OFF_WU1T);
  u16b* WU1t_b = WU1t_f + (size_t)512 * K1_;
  u16b* WU2t_f = (u16b*)(ws + OFF_WU2T);
  u16b* WU2t_b = WU2t_f + (size_t)512 * K2_;
  u16b* Wdt    = (u16b*)(ws + OFF_WDT);

  auto tr = [&](const float* src, int K, int N, u16b* dst, int stride, int k0) {
    int total = K * N;
    k_transpose_bf16<<<(total + 255) / 256, 256, 0, stream>>>(src, K, N, dst, stride, k0);
  };
  tr(Wf1, F_, H_, WU1t_f, K1_, 0);
  tr(Uf1, H_, H_, WU1t_f, K1_, F_);
  tr(Wf2, H_, H_, WU2t_f, K2_, 0);
  tr(Uf2, H_, H_, WU2t_f, K2_, H_);
  tr(Wb1, F_, H_, WU1t_b, K1_, 0);
  tr(Ub1, H_, H_, WU1t_b, K1_, F_);
  tr(Wb2, H_, H_, WU2t_b, K2_, 0);
  tr(Ub2, H_, H_, WU2t_b, K2_, H_);
  tr(Wd, H2_, NH_, Wdt, K2_, 0);

  {
    dim3 g(T_ * B_, 2);
    k_xprep<<<g, F_, 0, stream>>>(x1, x2, (u16b*)(ws + OFF_XB), (float*)(ws + OFF_MASK));
  }
  {
    size_t shmem = 128u * 1024u * 2u + 2u * 128u * 128u * 2u;  /* 256KB + 64KB = 320KB */
    k_rnn<<<4, 1024, shmem, stream>>>(ws, bf1, bf2, bb1, bb2);
  }
  k_bn<<<(2 * H2_ + 255) / 256, 256, 0, stream>>>(ws, gamma, beta);
  k_dense<<<2, 512, 0, stream>>>(ws, bd);
  k_cos<<<B_, 32, 0, stream>>>(ws, (float*)d_out);
}
